// DeformableFeatureAlignment_22960895164778
// MI455X (gfx1250) — matrix operand order repaired and checked by reading, not compiled, not run on hardware
//
#include <hip/hip_runtime.h>

// DeformableFeatureAlignment on gfx1250 (MI455X):
//  - all matmuls via v_wmma_f32_16x16x32_bf16 (bf16 in, f32 accumulate)
//  - weight panels staged LDS-image-identical in global memory and pulled
//    with the Tensor Data Mover (tensor_load_to_lds), double buffered,
//    synchronized with s_wait_tensorcnt.

#define B_    4
#define H_    64
#define W_    64
#define C_    256           // Cf = Cc = F
#define PX    (B_*H_*W_)    // 16384 pixel rows
#define DGRP  8
#define KTAP  9
#define OMC   (DGRP*3*KTAP) // 216
#define NOFF  (DGRP*2*KTAP) // 144

typedef __attribute__((ext_vector_type(16))) __bf16 v16bf;
typedef __attribute__((ext_vector_type(8)))  float  v8f;
typedef __attribute__((ext_vector_type(4)))  unsigned int u32x4;
typedef __attribute__((ext_vector_type(8)))  int          i32x8;
typedef __attribute__((ext_vector_type(4)))  int          i32x4;

union Frag16 { v16bf v; unsigned int u[8]; };

__device__ __forceinline__ unsigned short f2bf(float f) {
  unsigned int u = __float_as_uint(f);
  u += 0x7FFFu + ((u >> 16) & 1u);          // round-to-nearest-even
  return (unsigned short)(u >> 16);
}

__device__ __forceinline__ float sigmoidf(float x) {
  return 1.0f / (1.0f + __expf(-x));
}

// ---- WMMA tiling: block = 256 threads = 8 wave32, tile 64(M) x 256(N) ----
#define BM   64
#define BN   256
#define ASTR 34             // LDS row stride (ushorts) for A tile [BM][32]
#define BSTR 34             // LDS row stride (ushorts) for B^T tile [BN][32]
#define PPAN (BN*BSTR)      // 8704 ushorts = 17408 B per packed weight panel

// A fragment (16x32 bf16): lanes 0-15 M=0-15 K=0-15,
// lanes 16-31 M=0-15 K=16-31; 2 bf16 per VGPR.
__device__ __forceinline__ v16bf frag_a(const unsigned short* As, int lane) {
  const unsigned short* row = As + (lane & 15) * ASTR + ((lane & 16) ? 16 : 0);
  Frag16 f;
#pragma unroll
  for (int i = 0; i < 8; ++i) {
    f.u[i]     = *(const unsigned int*)(row + 2 * i);
  }
  return f.v;
}

// B fragment (32x16 bf16) from K-transposed LDS tile Bt[n][k].
// Contiguous k (same as A): lanes 0-15 N=lane K=0-15, lanes 16-31 N=lane-16 K=16-31.
__device__ __forceinline__ v16bf frag_b(const unsigned short* Bt, int lane) {
  const unsigned short* row = Bt + (lane & 15) * BSTR + ((lane & 16) ? 16 : 0);
  Frag16 f;
#pragma unroll
  for (int i = 0; i < 8; ++i) f.u[i] = *(const unsigned int*)(row + 2 * i);
  return f.v;
}

__device__ __forceinline__ v8f wmma_bf16(v16bf a, v16bf b, v8f c) {
  return __builtin_amdgcn_wmma_f32_16x16x32_bf16(false, a, false, b,
                                                 (short)0, c, false, false);
}

// TDM: copy one packed 17408-byte weight panel (LDS-image layout) into LDS.
// 1-D tensor: data_size=2B, tile_dim0 = tensor_dim0 = 8704 elements.
__device__ __forceinline__ void tdm_load_panel(const unsigned short* gsrc,
                                               const unsigned short* lds_dst) {
  unsigned long long ga = (unsigned long long)(uintptr_t)gsrc;
  unsigned galo = (unsigned)__builtin_amdgcn_readfirstlane((int)(ga & 0xFFFFFFFFu));
  unsigned gahi = (unsigned)__builtin_amdgcn_readfirstlane((int)(ga >> 32));
  unsigned loff = (unsigned)__builtin_amdgcn_readfirstlane(
      (int)(unsigned)(uintptr_t)lds_dst);      // flat LDS addr: low 32 = offset
  u32x4 g0;
  g0.x = 1u;                                   // count=1 valid descriptor
  g0.y = loff;                                 // lds_addr
  g0.z = galo;                                 // global_addr[31:0]
  g0.w = (gahi & 0x1FFFFFFu) | (2u << 30);     // global_addr[56:32] | type=2
  i32x8 g1;
  g1[0] = 0x00010000;                          // wg_mask=0, data_size=2B
  g1[1] = (int)((PPAN & 0xFFFFu) << 16);       // tensor_dim0[15:0]
  g1[2] = (int)((PPAN >> 16) | (1u << 16));    // tensor_dim0[31:16] | dim1=1
  g1[3] = (int)((unsigned)PPAN << 16);         // tile_dim0 = 8704
  g1[4] = 0;                                   // tile_dim1=0 (1-D), tile_dim2=0
  g1[5] = (int)PPAN;                           // tensor_dim0_stride
  g1[6] = 0;
  g1[7] = 0;
  i32x4 z4 = {0, 0, 0, 0};                     // groups 2/3 unused (<=2-D)
  i32x8 z8 = {0, 0, 0, 0, 0, 0, 0, 0};         // trailing arg (clang-23 form)
  __builtin_amdgcn_tensor_load_to_lds(g0, g1, z4, z4, z8, 0);
}

// ---------------- weight packing (fp32 -> bf16 LDS-image panels) -----------

// One block per 32-row K-panel: dst panel kp holds [n=0..255][kk=0..31] at
// n*BSTR+kk  (pad lanes left untouched; never read after the TDM blit).
__global__ void k_pack(const float* __restrict__ w, int ldn, int ncols, int col0,
                       unsigned short* __restrict__ dst) {
  int kp = blockIdx.x;
  unsigned short* p = dst + (size_t)kp * PPAN;
#pragma unroll
  for (int e = 0; e < 32; ++e) {
    int idx = threadIdx.x + e * 256;           // 8192 payload entries
    int n = idx >> 5, kk = idx & 31;
    int nc = col0 + n;
    float v = (nc < ncols) ? w[(size_t)(kp * 32 + kk) * ldn + nc] : 0.f;
    p[n * BSTR + kk] = f2bf(v);
  }
}

// ---------------- small prologue kernels ----------------

__global__ void k_gap(const float* __restrict__ fine, float* __restrict__ gap) {
  int b = blockIdx.x, c = threadIdx.x;
  const float* p = fine + (size_t)b * H_ * W_ * C_ + c;
  float s = 0.f;
  for (int i = 0; i < H_ * W_; ++i) s += p[(size_t)i * C_];
  gap[b * C_ + c] = s * (1.0f / (H_ * W_));
}

__global__ void k_gate(const float* __restrict__ gap,
                       const float* __restrict__ attend_w,
                       float* __restrict__ gate) {
  int b = blockIdx.x, f = threadIdx.x;
  const float* g = gap + b * C_;
  float s = 0.f;
  for (int c = 0; c < C_; ++c) s += g[c] * attend_w[c * C_ + f];
  gate[b * C_ + f] = 1.0f + sigmoidf(s);       // attn + 1
}

// bilinear upsample 32x32 -> 64x64, half-pixel centers (jax.image.resize)
__global__ void k_upsample(const float* __restrict__ coarse,
                           float* __restrict__ up) {
  int idx = blockIdx.x * 256 + threadIdx.x;
  if (idx >= B_ * H_ * W_ * C_) return;
  int c = idx & 255, x = (idx >> 8) & 63, y = (idx >> 14) & 63, b = idx >> 20;
  float fy = y * 0.5f - 0.25f, fx = x * 0.5f - 0.25f;
  float y0f = floorf(fy), x0f = floorf(fx);
  float ty = fy - y0f, tx = fx - x0f;
  int y0 = (int)y0f, x0 = (int)x0f;
  int y0c = min(max(y0, 0), 31), y1c = min(max(y0 + 1, 0), 31);
  int x0c = min(max(x0, 0), 31), x1c = min(max(x0 + 1, 0), 31);
  const float* cb = coarse + ((size_t)b * 32 * 32) * C_ + c;
  float v00 = cb[(size_t)(y0c * 32 + x0c) * C_];
  float v01 = cb[(size_t)(y0c * 32 + x1c) * C_];
  float v10 = cb[(size_t)(y1c * 32 + x0c) * C_];
  float v11 = cb[(size_t)(y1c * 32 + x1c) * C_];
  up[idx] = (1.f - ty) * ((1.f - tx) * v00 + tx * v01) +
            ty * ((1.f - tx) * v10 + tx * v11);
}

// --- shared per-K-step TDM sequence (issued by wave 0, double buffered) ----
#define TDM_STEP(kp, NK, bpack, Bt)                                           \
  do {                                                                        \
    if (ww == 0) {                                                            \
      if ((kp) + 1 < (NK)) {                                                  \
        tdm_load_panel((bpack) + (size_t)((kp) + 1) * PPAN,                   \
                       &Bt[((kp) + 1) & 1][0]);                               \
        __builtin_amdgcn_s_wait_tensorcnt(1);                                 \
      } else {                                                                \
        __builtin_amdgcn_s_wait_tensorcnt(0);                                 \
      }                                                                       \
    }                                                                         \
  } while (0)

// ---------------- WMMA GEMM kernels ----------------

// fine_cal = (fine * gate[b]) @ select_w   [16384x256]·[256x256]
__global__ void __launch_bounds__(256)
k_gemm_finecal(const float* __restrict__ fine, const float* __restrict__ gate,
               const unsigned short* __restrict__ bpack,
               float* __restrict__ fine_cal) {
  __shared__ __align__(16) unsigned short As[BM * ASTR];
  __shared__ __align__(16) unsigned short Bt[2][PPAN];
  const int tid = threadIdx.x, lane = tid & 31, ww = tid >> 5;
  const int wm = ww >> 1, wn = ww & 1;
  const int p0 = blockIdx.x * BM;
  const int NK = C_ / 32;
  v8f acc[8] = {};
  if (ww == 0) tdm_load_panel(bpack, &Bt[0][0]);
  for (int kp = 0; kp < NK; ++kp) {
    const int k0 = kp * 32;
#pragma unroll
    for (int e = 0; e < 8; ++e) {              // A tile: 64x32, gated fine
      int idx = tid + e * 256, r = idx >> 5, kk = idx & 31;
      int p = p0 + r, kg = k0 + kk;
      As[r * ASTR + kk] =
          f2bf(fine[(size_t)p * C_ + kg] * gate[(p >> 12) * C_ + kg]);
    }
    TDM_STEP(kp, NK, bpack, Bt);
    __syncthreads();
    const unsigned short* bt = &Bt[kp & 1][0];
    v16bf a = frag_a(As + wm * 16 * ASTR, lane);
#pragma unroll
    for (int j = 0; j < 8; ++j)
      acc[j] = wmma_bf16(a, frag_b(bt + (wn * 128 + j * 16) * BSTR, lane), acc[j]);
    __syncthreads();
  }
  const int m0 = p0 + wm * 16 + ((lane & 16) ? 8 : 0);
  const int n0 = wn * 128 + (lane & 15);
#pragma unroll
  for (int j = 0; j < 8; ++j)
#pragma unroll
    for (int r = 0; r < 8; ++r)
      fine_cal[(size_t)(m0 + r) * C_ + n0 + j * 16] = acc[j][r];
}

// align = concat(fine_cal, 2*coarse_up) @ offset_w  [16384x512]·[512x512] -> bf16
__global__ void __launch_bounds__(256)
k_gemm_align(const float* __restrict__ fine_cal, const float* __restrict__ coarse_up,
             const unsigned short* __restrict__ bpack_all,
             unsigned short* __restrict__ align_bf) {
  __shared__ __align__(16) unsigned short As[BM * ASTR];
  __shared__ __align__(16) unsigned short Bt[2][PPAN];
  const int tid = threadIdx.x, lane = tid & 31, ww = tid >> 5;
  const int wm = ww >> 1, wn = ww & 1;
  const int p0 = blockIdx.x * BM;
  const int col0 = blockIdx.y * BN;            // 512 output cols in 2 slabs
  const int NK = 512 / 32;
  const unsigned short* bpack = bpack_all + (size_t)blockIdx.y * NK * PPAN;
  v8f acc[8] = {};
  if (ww == 0) tdm_load_panel(bpack, &Bt[0][0]);
  for (int kp = 0; kp < NK; ++kp) {
    const int k0 = kp * 32;
#pragma unroll
    for (int e = 0; e < 8; ++e) {              // A tile: concat + 2x scale
      int idx = tid + e * 256, r = idx >> 5, kk = idx & 31;
      int p = p0 + r, kg = k0 + kk;
      float v = (kg < C_) ? fine_cal[(size_t)p * C_ + kg]
                          : 2.0f * coarse_up[(size_t)p * C_ + (kg - C_)];
      As[r * ASTR + kk] = f2bf(v);
    }
    TDM_STEP(kp, NK, bpack, Bt);
    __syncthreads();
    const unsigned short* bt = &Bt[kp & 1][0];
    v16bf a = frag_a(As + wm * 16 * ASTR, lane);
#pragma unroll
    for (int j = 0; j < 8; ++j)
      acc[j] = wmma_bf16(a, frag_b(bt + (wn * 128 + j * 16) * BSTR, lane), acc[j]);
    __syncthreads();
  }
  const int m0 = p0 + wm * 16 + ((lane & 16) ? 8 : 0);
  const int n0 = wn * 128 + (lane & 15);
#pragma unroll
  for (int j = 0; j < 8; ++j)
#pragma unroll
    for (int r = 0; r < 8; ++r)
      align_bf[(size_t)(m0 + r) * 512 + col0 + n0 + j * 16] = f2bf(acc[j][r]);
}

// om = conv3x3_same(align) + bias; mask channels (>=144) get sigmoid.
// im2col GEMM: M=16384, K=9*512=4608 (tap-major), N=216 (padded to 256).
__global__ void __launch_bounds__(256)
k_gemm_om(const unsigned short* __restrict__ align_bf,
          const unsigned short* __restrict__ bpack,
          const float* __restrict__ om_b, float* __restrict__ om) {
  __shared__ __align__(16) unsigned short As[BM * ASTR];
  __shared__ __align__(16) unsigned short Bt[2][PPAN];
  const int tid = threadIdx.x, lane = tid & 31, ww = tid >> 5;
  const int wm = ww >> 1, wn = ww & 1;
  const int p0 = blockIdx.x * BM;
  const int NK = (9 * 512) / 32;
  v8f acc[8] = {};
  if (ww == 0) tdm_load_panel(bpack, &Bt[0][0]);
  for (int kp = 0; kp < NK; ++kp) {
    const int k0 = kp * 32;
#pragma unroll
    for (int e = 0; e < 8; ++e) {              // zero-padded im2col A tile
      int idx = tid + e * 256, r = idx >> 5, kk = idx & 31;
      int p = p0 + r, kg = k0 + kk;
      int tap = kg >> 9, c = kg & 511;
      int y = (p >> 6) & 63, x = p & 63, b = p >> 12;
      int yy = y + tap / 3 - 1, xx = x + tap % 3 - 1;
      unsigned short v = 0;
      if ((unsigned)yy < 64u && (unsigned)xx < 64u)
        v = align_bf[(size_t)((b * 64 + yy) * 64 + xx) * 512 + c];
      As[r * ASTR + kk] = v;
    }
    TDM_STEP(kp, NK, bpack, Bt);
    __syncthreads();
    const unsigned short* bt = &Bt[kp & 1][0];
    v16bf a = frag_a(As + wm * 16 * ASTR, lane);
#pragma unroll
    for (int j = 0; j < 8; ++j)
      acc[j] = wmma_bf16(a, frag_b(bt + (wn * 128 + j * 16) * BSTR, lane), acc[j]);
    __syncthreads();
  }
  const int m0 = p0 + wm * 16 + ((lane & 16) ? 8 : 0);
  const int n0 = wn * 128 + (lane & 15);
#pragma unroll
  for (int j = 0; j < 8; ++j) {
    int n = n0 + j * 16;
    if (n < OMC) {
#pragma unroll
      for (int r = 0; r < 8; ++r) {
        float v = acc[j][r] + om_b[n];
        if (n >= NOFF) v = sigmoidf(v);        // modulation mask
        om[(size_t)(m0 + r) * OMC + n] = v;
      }
    }
  }
}

// DCNv2 + epilogue: out = relu(val @ dcn_w + dcn_b) + fine_cal.
// K = 9*256 tap-major; each 32-chunk = one (tap, deformable group).
__global__ void __launch_bounds__(256)
k_gemm_dcn(const float* __restrict__ coarse_up, const float* __restrict__ om,
           const unsigned short* __restrict__ bpack, const float* __restrict__ dcn_b,
           const float* __restrict__ fine_cal, float* __restrict__ out) {
  __shared__ __align__(16) unsigned short As[BM * ASTR];
  __shared__ __align__(16) unsigned short Bt[2][PPAN];
  const int tid = threadIdx.x, lane = tid & 31, ww = tid >> 5;
  const int wm = ww >> 1, wn = ww & 1;
  const int p0 = blockIdx.x * BM;
  const int NK = (9 * C_) / 32;
  v8f acc[8] = {};
  if (ww == 0) tdm_load_panel(bpack, &Bt[0][0]);
  for (int kp = 0; kp < NK; ++kp) {
    const int k0 = kp * 32;
    const int tap = k0 >> 8;                   // 256 channels per tap
    const int g   = (k0 & 255) >> 5;           // deformable group of this chunk
    const int dy  = tap / 3 - 1, dx = tap % 3 - 1;
#pragma unroll
    for (int e = 0; e < 8; ++e) {              // deformable-sampled A tile
      int idx = tid + e * 256, r = idx >> 5, kk = idx & 31;
      int p = p0 + r;
      int y = (p >> 6) & 63, x = p & 63, b = p >> 12;
      int c = (k0 & 255) + kk;
      const float* omp = om + (size_t)p * OMC;
      float offy = omp[g * 18 + tap * 2 + 0];
      float offx = omp[g * 18 + tap * 2 + 1];
      float msk  = omp[NOFF + g * 9 + tap];
      float yp = (float)(y + dy) + offy;
      float xp = (float)(x + dx) + offx;
      float y0f = floorf(yp), x0f = floorf(xp);
      float wy = yp - y0f, wx = xp - x0f;
      int y0 = (int)y0f, x0 = (int)x0f;
      const float* base = coarse_up + ((size_t)b * 64 * 64) * C_ + c;
      float v00 = ((unsigned)y0 < 64u && (unsigned)x0 < 64u)
                      ? base[(size_t)(y0 * 64 + x0) * C_] : 0.f;
      float v01 = ((unsigned)y0 < 64u && (unsigned)(x0 + 1) < 64u)
                      ? base[(size_t)(y0 * 64 + x0 + 1) * C_] : 0.f;
      float v10 = ((unsigned)(y0 + 1) < 64u && (unsigned)x0 < 64u)
                      ? base[(size_t)((y0 + 1) * 64 + x0) * C_] : 0.f;
      float v11 = ((unsigned)(y0 + 1) < 64u && (unsigned)(x0 + 1) < 64u)
                      ? base[(size_t)((y0 + 1) * 64 + x0 + 1) * C_] : 0.f;
      float v = (1.f - wy) * ((1.f - wx) * v00 + wx * v01) +
                wy * ((1.f - wx) * v10 + wx * v11);
      As[r * ASTR + kk] = f2bf(v * msk);
    }
    TDM_STEP(kp, NK, bpack, Bt);
    __syncthreads();
    const unsigned short* bt = &Bt[kp & 1][0];
    v16bf a = frag_a(As + wm * 16 * ASTR, lane);
#pragma unroll
    for (int j = 0; j < 8; ++j)
      acc[j] = wmma_bf16(a, frag_b(bt + (wn * 128 + j * 16) * BSTR, lane), acc[j]);
    __syncthreads();
  }
  const int m0 = p0 + wm * 16 + ((lane & 16) ? 8 : 0);
  const int n0 = wn * 128 + (lane & 15);
#pragma unroll
  for (int j = 0; j < 8; ++j) {
    int n = n0 + j * 16;
#pragma unroll
    for (int r = 0; r < 8; ++r) {
      size_t o = (size_t)(m0 + r) * C_ + n;
      float v = acc[j][r] + dcn_b[n];
      out[o] = fmaxf(v, 0.f) + fine_cal[o];    // relu(aligned) + fine_cal
    }
  }
}

// ---------------- launcher ----------------

extern "C" void kernel_launch(void* const* d_in, const int* in_sizes, int n_in,
                              void* d_out, int out_size, void* d_ws, size_t ws_size,
                              hipStream_t stream) {
  const float* fine     = (const float*)d_in[0];
  const float* coarse   = (const float*)d_in[1];
  const float* attend_w = (const float*)d_in[2];
  const float* select_w = (const float*)d_in[3];
  const float* offset_w = (const float*)d_in[4];
  const float* om_w     = (const float*)d_in[5];
  const float* om_b     = (const float*)d_in[6];
  const float* dcn_w    = (const float*)d_in[7];
  const float* dcn_b    = (const float*)d_in[8];
  float* out = (float*)d_out;

  char* ws = (char*)d_ws;
  float*          gap       = (float*)(ws + 0);                   //   4 KB
  float*          gate      = (float*)(ws + 4096);                //   4 KB
  float*          coarse_up = (float*)(ws + 8192);                //  16 MB
  float*          fine_cal  = (float*)(ws + 8192 + 16777216ull);  //  16 MB
  unsigned short* align_bf  = (unsigned short*)(ws + 8192 + 2ull * 16777216); // 16 MB
  float*          om        = (float*)(ws + 8192 + 3ull * 16777216);          // 13.5 MB
  // bf16 LDS-image weight panels (16-byte aligned offsets)
  unsigned short* pk_sel = (unsigned short*)(ws + 64495616ull);   //   8 panels
  unsigned short* pk_off = (unsigned short*)(ws + 64634880ull);   //  32 panels
  unsigned short* pk_omw = (unsigned short*)(ws + 65191936ull);   // 144 panels
  unsigned short* pk_dcn = (unsigned short*)(ws + 67698688ull);   //  72 panels

  // weight packing (runs every call; deterministic)
  k_pack<<<8,   256, 0, stream>>>(select_w, 256, 256, 0,   pk_sel);
  k_pack<<<16,  256, 0, stream>>>(offset_w, 512, 512, 0,   pk_off);
  k_pack<<<16,  256, 0, stream>>>(offset_w, 512, 512, 256, pk_off + 16ull * PPAN);
  k_pack<<<144, 256, 0, stream>>>(om_w,     216, 216, 0,   pk_omw);
  k_pack<<<72,  256, 0, stream>>>(dcn_w,    256, 256, 0,   pk_dcn);

  k_gap<<<B_, C_, 0, stream>>>(fine, gap);
  k_gate<<<B_, C_, 0, stream>>>(gap, attend_w, gate);
  k_upsample<<<(B_ * H_ * W_ * C_ + 255) / 256, 256, 0, stream>>>(coarse, coarse_up);
  k_gemm_finecal<<<dim3(PX / BM), 256, 0, stream>>>(fine, gate, pk_sel, fine_cal);
  k_gemm_align<<<dim3(PX / BM, 2), 256, 0, stream>>>(fine_cal, coarse_up, pk_off, align_bf);
  k_gemm_om<<<dim3(PX / BM), 256, 0, stream>>>(align_bf, pk_omw, om_b, om);
  k_gemm_dcn<<<dim3(PX / BM), 256, 0, stream>>>(coarse_up, om, pk_dcn, dcn_b, fine_cal, out);
}
